// UL_CE_81020263072086
// MI455X (gfx1250) — matrix operand order repaired and checked by reading, not compiled, not run on hardware
//
#include <hip/hip_runtime.h>
#include <math.h>

#define B_SZ 16
#define NCC  64
#define NRR  4
#define RFF  4
#define LUL  8
#define MA   256
#define NA   1024

typedef __attribute__((ext_vector_type(16))) __bf16 v16bf;
typedef __attribute__((ext_vector_type(8)))  __bf16 v8bf;
typedef __attribute__((ext_vector_type(8)))  float  v8f;

__device__ __forceinline__ float fm_of(int c) {
    return ((float)(c + 1) - 32.5f) * (1.0e9f / 64.0f) + 1.0e11f;
}
__device__ __forceinline__ float sigmoidf_(float x) { return 1.0f / (1.0f + expf(-x)); }

// Build a v16bf WMMA fragment from two 16-byte LDS chunks.
__device__ __forceinline__ v16bf ldfrag(const __bf16* p0, const __bf16* p1) {
    union { struct { v8bf lo, hi; } s; v16bf v; } u;
    u.s.lo = *(const v8bf*)p0;
    u.s.hi = *(const v8bf*)p1;
    return u.v;
}

// ---------------- theta: [L][Nc][N] complex ----------------
// BIT=1 => exp(i*floor(2P)*pi) = (-1)^floor(2P): pure sign.
__global__ void theta_kernel(const float* __restrict__ Phase1,
                             const float* __restrict__ Phase2,
                             const float* __restrict__ PhaseTDD,
                             float* __restrict__ thRe, float* __restrict__ thIm) {
    int idx = blockIdx.x * 256 + threadIdx.x;   // 8*64*1024 total
    int n = idx & 1023;
    int c = (idx >> 10) & 63;
    int l = idx >> 16;
    int k1 = (int)floorf(2.0f * Phase1[l * NA + n]);
    int k2 = (int)floorf(2.0f * Phase2[l * NA + n]);
    float sgn = ((k1 + k2) & 1) ? -1.0f : 1.0f;
    float a1 = 5e-9f * sigmoidf_(PhaseTDD[l * 16 + (n >> 6)]);
    float ang = 6.28318530718f * fm_of(c) * a1;
    float s, co; sincosf(ang, &s, &co);
    thRe[idx] = sgn * co;
    thIm[idx] = sgn * s;
}

// ---------------- F: [L][Nc][M][RF] complex ----------------
__global__ void f_kernel(const float* __restrict__ S_analog,
                         const float* __restrict__ S_TDD,
                         float* __restrict__ FRe, float* __restrict__ FIm) {
    int idx = blockIdx.x * 256 + threadIdx.x;   // 8*64*256*4 total
    int f = idx & 3;
    int m = (idx >> 2) & 255;
    int c = (idx >> 10) & 63;
    int l = idx >> 16;
    float bdel = 5e-9f * sigmoidf_(S_TDD[((l * NCC + c) * 16 + (m >> 4)) * 4 + f]);
    float ang = S_analog[idx] + 6.28318530718f * fm_of(c) * bdel;
    float s, co; sincosf(ang, &s, &co);
    FRe[idx] = co * 0.0625f;   // 1/sqrt(256)
    FIm[idx] = s * 0.0625f;
}

// ---------------- fused complex GEMM (WMMA bf16) + y0 + noise + x layout ----
// grid (64 c, 8 rowBlocks); block 256 = 8 waves; wave = 16 rows x 128 cols.
// Dynamic LDS layout (bytes):
//   staging: Ar[64*32]bf16 @0, Ai @4096, Btr[256*32]bf16 @8192, Bti @24576, Btn @40960
//   epilogue (reuse): aRe[64*256]f32 @0, aIm @65536, pow[16]f32 @131072
#define GEMM_LDS_BYTES 131136
__global__ __launch_bounds__(256, 1) void gemm_kernel(
    const float* __restrict__ HBRre, const float* __restrict__ HBRim,
    const float* __restrict__ HRUre, const float* __restrict__ HRUim,
    const float* __restrict__ HBUre, const float* __restrict__ HBUim,
    const float* __restrict__ nzre,  const float* __restrict__ nzim,
    const float* __restrict__ thRe,  const float* __restrict__ thIm,
    const float* __restrict__ FRe,   const float* __restrict__ FIm,
    float* __restrict__ xout) {
    extern __shared__ char smem[];
    __bf16* Ar  = (__bf16*)(smem);
    __bf16* Ai  = (__bf16*)(smem + 4096);
    __bf16* Btr = (__bf16*)(smem + 8192);
    __bf16* Bti = (__bf16*)(smem + 24576);
    __bf16* Btn = (__bf16*)(smem + 40960);
    float*  aRe = (float*)(smem);
    float*  aIm = (float*)(smem + 65536);
    float*  powLds = (float*)(smem + 131072);

    const int c   = blockIdx.x;
    const int rb  = blockIdx.y;
    const int tid = threadIdx.x;
    const int lane = tid & 31;
    const int w    = tid >> 5;
    const int rowTile = w & 3;    // 4 row tiles of 16
    const int colHalf = w >> 2;   // 2 col halves of 128

    // Per-thread A-staging assignment: one row, 8 k's.
    const int fillRow = tid >> 2;
    const int fillKq  = (tid & 3) << 3;
    {
        // nothing
    }
    int R0 = rb * 64 + fillRow;
    int fb = R0 >> 5, fl = (R0 >> 2) & 7, fr_ = R0 & 3;
    const float* hruRe = HRUre + (((size_t)fb * NCC + c) * NRR + fr_) * NA;
    const float* hruIm = HRUim + (((size_t)fb * NCC + c) * NRR + fr_) * NA;
    const float* thReR = thRe + ((size_t)fl * NCC + c) * NA;
    const float* thImR = thIm + ((size_t)fl * NCC + c) * NA;

    v8f accRe[8], accIm[8];
#pragma unroll
    for (int t = 0; t < 8; ++t) { accRe[t] = (v8f){}; accIm[t] = (v8f){}; }

    const int arow  = rowTile * 16 + (lane & 15);
    const int abase = (lane < 16) ? 0 : 8;     // A: 16-bit 16x32 lane layout
    const int bbase = (lane < 16) ? 0 : 8;     // B: 16-bit 32x16 lane layout

#pragma unroll 1
    for (int k0 = 0; k0 < NA; k0 += 32) {
        // stage A = H_RU * theta (complex), bf16
#pragma unroll
        for (int j = 0; j < 8; ++j) {
            int kg = k0 + fillKq + j;
            float hr = hruRe[kg], hi = hruIm[kg];
            float tr = thReR[kg], ti = thImR[kg];
            Ar[fillRow * 32 + fillKq + j] = (__bf16)(hr * tr - hi * ti);
            Ai[fillRow * 32 + fillKq + j] = (__bf16)(hr * ti + hi * tr);
        }
        // stage B = H_BR[c, k0..k0+32, :] transposed to [m][k], bf16 (+ neg imag)
        {
            const float* bre = HBRre + ((size_t)c * NA + k0) * MA + tid;
            const float* bim = HBRim + ((size_t)c * NA + k0) * MA + tid;
#pragma unroll 4
            for (int k = 0; k < 32; ++k) {
                float br = bre[(size_t)k * MA], bi = bim[(size_t)k * MA];
                Btr[tid * 32 + k] = (__bf16)br;
                Bti[tid * 32 + k] = (__bf16)bi;
                Btn[tid * 32 + k] = (__bf16)(-bi);
            }
        }
        __syncthreads();

        v16bf ar = ldfrag(&Ar[arow * 32 + abase], &Ar[arow * 32 + abase + 16]);
        v16bf ai = ldfrag(&Ai[arow * 32 + abase], &Ai[arow * 32 + abase + 16]);
#pragma unroll
        for (int t = 0; t < 8; ++t) {
            int ncol = colHalf * 128 + t * 16 + (lane & 15);
            v16bf br = ldfrag(&Btr[ncol * 32 + bbase], &Btr[ncol * 32 + bbase + 16]);
            v16bf bi = ldfrag(&Bti[ncol * 32 + bbase], &Bti[ncol * 32 + bbase + 16]);
            v16bf bn = ldfrag(&Btn[ncol * 32 + bbase], &Btn[ncol * 32 + bbase + 16]);
            accRe[t] = __builtin_amdgcn_wmma_f32_16x16x32_bf16(false, ar, false, br, (short)0, accRe[t], false, false);
            accRe[t] = __builtin_amdgcn_wmma_f32_16x16x32_bf16(false, ai, false, bn, (short)0, accRe[t], false, false);
            accIm[t] = __builtin_amdgcn_wmma_f32_16x16x32_bf16(false, ar, false, bi, (short)0, accIm[t], false, false);
            accIm[t] = __builtin_amdgcn_wmma_f32_16x16x32_bf16(false, ai, false, br, (short)0, accIm[t], false, false);
        }
        __syncthreads();
    }

    // epilogue: a = acc + H_BU into LDS (f32)
    const int laneN  = lane & 15;
    const int rowHi  = (lane >> 4) * 8;
#pragma unroll
    for (int t = 0; t < 8; ++t) {
        int col = colHalf * 128 + t * 16 + laneN;
#pragma unroll
        for (int v = 0; v < 8; ++v) {
            int rowLoc = rowTile * 16 + rowHi + v;
            int R = rb * 64 + rowLoc;
            int b = R >> 5, r = R & 3;
            size_t gi = (((size_t)b * NCC + c) * NRR + r) * MA + col;
            aRe[rowLoc * 256 + col] = accRe[t][v] + HBUre[gi];
            aIm[rowLoc * 256 + col] = accIm[t][v] + HBUim[gi];
        }
    }
    if (tid < 16) powLds[tid] = 0.0f;
    __syncthreads();

    // y0 = a @ F, power, noise, scatter into x[b][c][r*4+f][l / l+8]
    {
        int rowLoc = tid >> 2, f = tid & 3;
        int R = rb * 64 + rowLoc;
        int b = R >> 5, l = (R >> 2) & 7, r = R & 3;
        const float* fre = FRe + (((size_t)l * NCC + c) * MA) * RFF + f;
        const float* fim = FIm + (((size_t)l * NCC + c) * MA) * RFF + f;
        float y0r = 0.f, y0i = 0.f;
        for (int m = 0; m < MA; ++m) {
            float av = aRe[rowLoc * 256 + m], bv = aIm[rowLoc * 256 + m];
            float Frv = fre[m * 4], Fiv = fim[m * 4];
            y0r += av * Frv - bv * Fiv;
            y0i += av * Fiv + bv * Frv;
        }
        atomicAdd(&powLds[rowLoc >> 2], y0r * y0r + y0i * y0i);
        __syncthreads();
        float p = powLds[rowLoc >> 2];
        float sc = sqrtf(p * 0.1f * 0.5f);   // snr_lin=0.1
        size_t ni = ((((size_t)b * LUL + l) * NCC + c) * NRR + r) * RFF + f;
        float yr = y0r + nzre[ni] * sc;
        float yi = y0i + nzim[ni] * sc;
        size_t xo = (((size_t)b * NCC + c) * 16 + (r * 4 + f)) * 16 + l;
        xout[xo]     = yr;
        xout[xo + 8] = yi;
    }
}

// ---------------- PSA: one block per batch ----------------
#define PSA_LDS_BYTES 69184
__global__ __launch_bounds__(256, 1) void psa_kernel(
    const float* __restrict__ xin,
    const float* __restrict__ chwv_w, const float* __restrict__ chwv_b,
    const float* __restrict__ chwq_w, const float* __restrict__ chwq_b,
    const float* __restrict__ chwz_w, const float* __restrict__ chwz_b,
    const float* __restrict__ lnw,    const float* __restrict__ lnb,
    const float* __restrict__ spwv_w, const float* __restrict__ spwv_b,
    const float* __restrict__ spwq_w, const float* __restrict__ spwq_b,
    float* __restrict__ out) {
    extern __shared__ char smem[];
    float* xs   = (float*)(smem);           // 64*256
    float* q    = (float*)(smem + 65536);   // 256
    float* red  = (float*)(smem + 66560);   // 256
    float* xq   = (float*)(smem + 67584);   // 64
    float* xbar = (float*)(smem + 67840);   // 64
    float* cwz  = (float*)(smem + 68096);   // 32
    float* slog = (float*)(smem + 68224);   // 32
    float* zbuf = (float*)(smem + 68352);   // 64
    float* chw  = (float*)(smem + 68608);   // 64
    float* wcomb= (float*)(smem + 68864);   // 64
    float* scal = (float*)(smem + 69120);   // 16

    const int b = blockIdx.x, tid = threadIdx.x, n = tid;
    for (int i = tid; i < 64 * 256; i += 256) xs[i] = xin[(size_t)b * 16384 + i];
    __syncthreads();

    // channel-branch query softmax over hw
    float lg = chwq_b[0];
    for (int cc = 0; cc < 64; ++cc) lg += chwq_w[cc] * xs[cc * 256 + n];
    red[tid] = lg; __syncthreads();
    for (int s = 128; s > 0; s >>= 1) { if (tid < s) red[tid] = fmaxf(red[tid], red[tid + s]); __syncthreads(); }
    float mx = red[0]; __syncthreads();
    float e = expf(lg - mx);
    red[tid] = e; __syncthreads();
    for (int s = 128; s > 0; s >>= 1) { if (tid < s) red[tid] += red[tid + s]; __syncthreads(); }
    q[tid] = e / red[0];
    __syncthreads();

    if (tid < 64) {
        float sq = 0.f, sb = 0.f;
        for (int m = 0; m < 256; ++m) { float xv = xs[tid * 256 + m]; sq += xv * q[m]; sb += xv; }
        xq[tid] = sq; xbar[tid] = sb * (1.0f / 256.0f);
    }
    __syncthreads();

    if (tid < 32) {
        float s = chwv_b[tid];
        for (int cc = 0; cc < 64; ++cc) s += chwv_w[tid * 64 + cc] * xq[cc];
        cwz[tid] = s;
    } else if (tid < 64) {
        int k = tid - 32;
        float s = spwq_b[k];
        for (int cc = 0; cc < 64; ++cc) s += spwq_w[k * 64 + cc] * xbar[cc];
        slog[k] = s;
    }
    __syncthreads();

    float zc = 0.f;
    if (tid < 64) {
        zc = chwz_b[tid];
        for (int k = 0; k < 32; ++k) zc += chwz_w[tid * 32 + k] * cwz[k];
        zbuf[tid] = zc;
    }
    if (tid == 64) {   // spatial softmax over 32 (serial, tiny)
        float m2 = slog[0];
        for (int k = 1; k < 32; ++k) m2 = fmaxf(m2, slog[k]);
        float ss = 0.f;
        for (int k = 0; k < 32; ++k) { float ev = expf(slog[k] - m2); slog[k] = ev; ss += ev; }
        float inv = 1.0f / ss;
        for (int k = 0; k < 32; ++k) slog[k] *= inv;
    }
    __syncthreads();

    // LayerNorm over 64 channels
    red[tid] = (tid < 64) ? zbuf[tid] : 0.f; __syncthreads();
    for (int s = 32; s > 0; s >>= 1) { if (tid < s) red[tid] += red[tid + s]; __syncthreads(); }
    float mu = red[0] * (1.0f / 64.0f); __syncthreads();
    float d = (tid < 64) ? (zbuf[tid] - mu) : 0.f;
    red[tid] = d * d; __syncthreads();
    for (int s = 32; s > 0; s >>= 1) { if (tid < s) red[tid] += red[tid + s]; __syncthreads(); }
    float var = red[0] * (1.0f / 64.0f);
    if (tid < 64) {
        float zn = d * rsqrtf(var + 1e-5f) * lnw[tid] + lnb[tid];
        chw[tid] = 1.0f / (1.0f + expf(-zn));
        float wc = 0.f;
        for (int k = 0; k < 32; ++k) wc += slog[k] * spwv_w[k * 64 + tid];
        wcomb[tid] = wc;
    }
    if (tid == 0) {
        float bc = 0.f;
        for (int k = 0; k < 32; ++k) bc += slog[k] * spwv_b[k];
        scal[0] = bc;
    }
    __syncthreads();

    float swz = scal[0];
    for (int cc = 0; cc < 64; ++cc) swz += wcomb[cc] * xs[cc * 256 + n];
    float spw = 1.0f / (1.0f + expf(-swz));
    for (int cc = 0; cc < 64; ++cc)
        out[((size_t)b * 64 + cc) * 256 + n] = (spw + chw[cc]) * xs[cc * 256 + n];
}

extern "C" void kernel_launch(void* const* d_in, const int* in_sizes, int n_in,
                              void* d_out, int out_size, void* d_ws, size_t ws_size,
                              hipStream_t stream) {
    (void)in_sizes; (void)n_in; (void)out_size; (void)ws_size;
    const float* HBRre  = (const float*)d_in[0];
    const float* HBRim  = (const float*)d_in[1];
    const float* HRUre  = (const float*)d_in[2];
    const float* HRUim  = (const float*)d_in[3];
    const float* HBUre  = (const float*)d_in[4];
    const float* HBUim  = (const float*)d_in[5];
    const float* nzre   = (const float*)d_in[6];
    const float* nzim   = (const float*)d_in[7];
    const float* Phase1 = (const float*)d_in[8];
    const float* Phase2 = (const float*)d_in[9];
    const float* PhaseTDD = (const float*)d_in[10];
    const float* S_analog = (const float*)d_in[11];
    const float* S_TDD    = (const float*)d_in[12];
    const float* chwv_w = (const float*)d_in[13];
    const float* chwv_b = (const float*)d_in[14];
    const float* chwq_w = (const float*)d_in[15];
    const float* chwq_b = (const float*)d_in[16];
    const float* chwz_w = (const float*)d_in[17];
    const float* chwz_b = (const float*)d_in[18];
    const float* lnw    = (const float*)d_in[19];
    const float* lnb    = (const float*)d_in[20];
    const float* spwv_w = (const float*)d_in[21];
    const float* spwv_b = (const float*)d_in[22];
    const float* spwq_w = (const float*)d_in[23];
    const float* spwq_b = (const float*)d_in[24];

    float* ws   = (float*)d_ws;
    float* thRe = ws;
    float* thIm = ws + 524288;
    float* FRe  = ws + 1048576;
    float* FIm  = ws + 1572864;
    float* x    = ws + 2097152;   // 262144 floats

    theta_kernel<<<2048, 256, 0, stream>>>(Phase1, Phase2, PhaseTDD, thRe, thIm);
    f_kernel<<<2048, 256, 0, stream>>>(S_analog, S_TDD, FRe, FIm);

    hipFuncSetAttribute(reinterpret_cast<const void*>(gemm_kernel),
                        hipFuncAttributeMaxDynamicSharedMemorySize, GEMM_LDS_BYTES);
    gemm_kernel<<<dim3(64, 8), 256, GEMM_LDS_BYTES, stream>>>(
        HBRre, HBRim, HRUre, HRUim, HBUre, HBUim, nzre, nzim,
        thRe, thIm, FRe, FIm, x);

    hipFuncSetAttribute(reinterpret_cast<const void*>(psa_kernel),
                        hipFuncAttributeMaxDynamicSharedMemorySize, PSA_LDS_BYTES);
    psa_kernel<<<16, 256, PSA_LDS_BYTES, stream>>>(
        x, chwv_w, chwv_b, chwq_w, chwq_b, chwz_w, chwz_b,
        lnw, lnb, spwv_w, spwv_b, spwq_w, spwq_b, (float*)d_out);
}
